// ProbSparseSelfAttention_64759516889436
// MI455X (gfx1250) — matrix operand order repaired and checked by reading, not compiled, not run on hardware
//
#include <hip/hip_runtime.h>
#include <hip/hip_bf16.h>

typedef _Float16 half_t;
typedef __attribute__((ext_vector_type(16))) _Float16 v16h;
typedef __attribute__((ext_vector_type(8)))  _Float16 v8h;
typedef __attribute__((ext_vector_type(8)))  float    v8f;

#define BDIM 2
#define LSEQ 4096
#define DMOD 1024
#define NH   16
#define HD   64
#define BHT  32
#define TOPU 819
#define SELP 832

union AFrag { v16h v; v8h h[2]; };

__device__ __forceinline__ v8f zero_v8f() {
  v8f z;
#pragma unroll
  for (int i = 0; i < 8; ++i) z[i] = 0.0f;
  return z;
}

__device__ __forceinline__ v8f wmma_f16(const AFrag& a, const AFrag& b, v8f c) {
  return __builtin_amdgcn_wmma_f32_16x16x32_f16(false, a.v, false, b.v,
                                                (short)0, c, false, false);
}

// CDNA5 async memory->LDS DMA (VGPR-bypassing), GVS mode:
// LDS destination is the byte address in VDST; source is SADDR base plus VADDR byte offset.
__device__ __forceinline__ void async_ld16(unsigned lds_addr,
                                           const half_t* gbase,
                                           unsigned byte_off) {
  asm volatile("global_load_async_to_lds_b128 %0, %1, %2"
               :: "v"(lds_addr), "v"(byte_off), "s"(gbase) : "memory");
}
// ASYNCcnt completes in-order per wave: <=6 means the previous 6-load batch done.
__device__ __forceinline__ void wait_async0() {
  asm volatile("s_wait_asynccnt 0" ::: "memory");
}
__device__ __forceinline__ void wait_async6() {
  asm volatile("s_wait_asynccnt 6" ::: "memory");
}
__device__ __forceinline__ unsigned lds_off(const void* p) {
  return (unsigned)(unsigned long long)p;   // generic addr[31:0] = LDS offset
}

// ---------------------------------------------------------------------------
// f32 -> f16 bulk convert (8 elems / thread)
// ---------------------------------------------------------------------------
__global__ __launch_bounds__(256) void k_cvt(const float* __restrict__ src,
                                             half_t* __restrict__ dst) {
  const int i = (blockIdx.x * 256 + threadIdx.x) * 8;
  float4 a = *(const float4*)(src + i);
  float4 b = *(const float4*)(src + i + 4);
  v8h o;
  o[0] = (half_t)a.x; o[1] = (half_t)a.y; o[2] = (half_t)a.z; o[3] = (half_t)a.w;
  o[4] = (half_t)b.x; o[5] = (half_t)b.y; o[6] = (half_t)b.z; o[7] = (half_t)b.w;
  *(v8h*)(dst + i) = o;
}

// ---------------------------------------------------------------------------
// QKV projection GEMM (f16 x f16 -> f32 acc), double-buffered async DMA.
// Block tile 128(m) x 64(n) x 32(k); 4 waves, wave tile 32 x 64 (8 WMMA/step).
// Q,K out as [BH][L][64] f16; V transposed out as [BH][64][L] f16.
// ---------------------------------------------------------------------------
__global__ __launch_bounds__(128) void k_proj_qkv(
    const half_t* __restrict__ xh,
    const half_t* __restrict__ Whq, const half_t* __restrict__ Whk,
    const half_t* __restrict__ Whv,
    const float* __restrict__ bq, const float* __restrict__ bk,
    const float* __restrict__ bv,
    half_t* __restrict__ Qh, half_t* __restrict__ Kh, half_t* __restrict__ Vth)
{
  __shared__ __align__(16) half_t As[2][128][40];
  __shared__ __align__(16) half_t Bs[2][64][40];
  const int which = blockIdx.z;
  const half_t* __restrict__ W  = (which == 0) ? Whq : (which == 1) ? Whk : Whv;
  const float* __restrict__ bias = (which == 0) ? bq : (which == 1) ? bk : bv;
  const int n0 = blockIdx.x * 64;
  const int m0 = blockIdx.y * 128;
  const int t = threadIdx.x;
  const int lane = t & 31, wave = t >> 5;
  const int ln = lane & 15, hf = lane >> 4;
  const int br = t >> 1, bc = (t & 1) * 2;

  v8f acc[2][4];
#pragma unroll
  for (int i = 0; i < 2; ++i)
#pragma unroll
    for (int j = 0; j < 4; ++j) acc[i][j] = zero_v8f();

  unsigned aL[2], bL[2];
  aL[0] = lds_off(&As[0][0][0]); aL[1] = lds_off(&As[1][0][0]);
  bL[0] = lds_off(&Bs[0][0][0]); bL[1] = lds_off(&Bs[1][0][0]);
  const unsigned aLoc = (unsigned)(t * 40) * 2u;
  const unsigned bLoc = (unsigned)(br * 40 + bc * 8) * 2u;
  const unsigned aRow0 = (unsigned)((m0 + t) * DMOD) * 2u;
  const unsigned bRow0 = (unsigned)((n0 + br) * DMOD + bc * 8) * 2u;

  // prologue: stage k0 = 0 into buffer 0
#pragma unroll
  for (int j = 0; j < 4; ++j) async_ld16(aL[0] + aLoc + j * 16u, xh, aRow0 + j * 16u);
#pragma unroll
  for (int j = 0; j < 2; ++j) async_ld16(bL[0] + bLoc + j * 16u, W, bRow0 + j * 16u);

#pragma unroll 2
  for (int it = 0; it < DMOD / 32; ++it) {
    const int cur = it & 1;
    if (it + 1 < DMOD / 32) {           // stage next k-tile into other buffer
      const unsigned koff = (unsigned)((it + 1) * 32) * 2u;
#pragma unroll
      for (int j = 0; j < 4; ++j)
        async_ld16(aL[1 - cur] + aLoc + j * 16u, xh, aRow0 + koff + j * 16u);
#pragma unroll
      for (int j = 0; j < 2; ++j)
        async_ld16(bL[1 - cur] + bLoc + j * 16u, W, bRow0 + koff + j * 16u);
      wait_async6();                    // current buffer's batch retired
    } else {
      wait_async0();
    }
    __syncthreads();                    // all waves' DMA into 'cur' visible

    AFrag af[2];
#pragma unroll
    for (int ms = 0; ms < 2; ++ms) {
      const int am = wave * 32 + ms * 16 + ln;
      af[ms].h[0] = *(const v8h*)&As[cur][am][hf * 8];   // A: K=e+8*hf | 16+e'+8*hf
      af[ms].h[1] = *(const v8h*)&As[cur][am][16 + hf * 8];
    }
#pragma unroll
    for (int tt = 0; tt < 4; ++tt) {
      AFrag bf;
      const int bn = tt * 16 + ln;
      bf.h[0] = *(const v8h*)&Bs[cur][bn][hf * 8];       // B: K=e+8*hf | 16+e'+8*hf
      bf.h[1] = *(const v8h*)&Bs[cur][bn][16 + hf * 8];
      acc[0][tt] = wmma_f16(af[0], bf, acc[0][tt]);
      acc[1][tt] = wmma_f16(af[1], bf, acc[1][tt]);
    }
    __syncthreads();                    // reads done before buffer reuse
  }

#pragma unroll
  for (int ms = 0; ms < 2; ++ms)
#pragma unroll
    for (int tt = 0; tt < 4; ++tt) {
      const int c = n0 + tt * 16 + ln;
      const float bcv = bias[c];
      const int h = c >> 6, d = c & 63;
#pragma unroll
      for (int v = 0; v < 8; ++v) {
        const int rm = m0 + wave * 32 + ms * 16 + v + 8 * hf;  // C: m=v+8*hf
        const int bb = rm >> 12, l = rm & (LSEQ - 1);
        const half_t hv = (half_t)(acc[ms][tt][v] + bcv);
        if (which == 2)
          Vth[(((size_t)bb * NH + h) * HD + d) * LSEQ + l] = hv;
        else if (which == 0)
          Qh[(((size_t)bb * NH + h) * LSEQ + l) * HD + d] = hv;
        else
          Kh[(((size_t)bb * NH + h) * LSEQ + l) * HD + d] = hv;
      }
    }
}

// ---------------------------------------------------------------------------
// Output projection: out = out_full(f16) @ Wo^T + bo (f32), double-buffered.
// ---------------------------------------------------------------------------
__global__ __launch_bounds__(128) void k_out_proj(
    const half_t* __restrict__ OF, const half_t* __restrict__ Who,
    const float* __restrict__ bo, float* __restrict__ out)
{
  __shared__ __align__(16) half_t As[2][128][40];
  __shared__ __align__(16) half_t Bs[2][64][40];
  const int n0 = blockIdx.x * 64;
  const int m0 = blockIdx.y * 128;
  const int t = threadIdx.x;
  const int lane = t & 31, wave = t >> 5;
  const int ln = lane & 15, hf = lane >> 4;
  const int br = t >> 1, bc = (t & 1) * 2;

  v8f acc[2][4];
#pragma unroll
  for (int i = 0; i < 2; ++i)
#pragma unroll
    for (int j = 0; j < 4; ++j) acc[i][j] = zero_v8f();

  unsigned aL[2], bL[2];
  aL[0] = lds_off(&As[0][0][0]); aL[1] = lds_off(&As[1][0][0]);
  bL[0] = lds_off(&Bs[0][0][0]); bL[1] = lds_off(&Bs[1][0][0]);
  const unsigned aLoc = (unsigned)(t * 40) * 2u;
  const unsigned bLoc = (unsigned)(br * 40 + bc * 8) * 2u;
  const unsigned aRow0 = (unsigned)((m0 + t) * DMOD) * 2u;
  const unsigned bRow0 = (unsigned)((n0 + br) * DMOD + bc * 8) * 2u;

#pragma unroll
  for (int j = 0; j < 4; ++j) async_ld16(aL[0] + aLoc + j * 16u, OF, aRow0 + j * 16u);
#pragma unroll
  for (int j = 0; j < 2; ++j) async_ld16(bL[0] + bLoc + j * 16u, Who, bRow0 + j * 16u);

#pragma unroll 2
  for (int it = 0; it < DMOD / 32; ++it) {
    const int cur = it & 1;
    if (it + 1 < DMOD / 32) {
      const unsigned koff = (unsigned)((it + 1) * 32) * 2u;
#pragma unroll
      for (int j = 0; j < 4; ++j)
        async_ld16(aL[1 - cur] + aLoc + j * 16u, OF, aRow0 + koff + j * 16u);
#pragma unroll
      for (int j = 0; j < 2; ++j)
        async_ld16(bL[1 - cur] + bLoc + j * 16u, Who, bRow0 + koff + j * 16u);
      wait_async6();
    } else {
      wait_async0();
    }
    __syncthreads();

    AFrag af[2];
#pragma unroll
    for (int ms = 0; ms < 2; ++ms) {
      const int am = wave * 32 + ms * 16 + ln;
      af[ms].h[0] = *(const v8h*)&As[cur][am][hf * 8];
      af[ms].h[1] = *(const v8h*)&As[cur][am][16 + hf * 8];
    }
#pragma unroll
    for (int tt = 0; tt < 4; ++tt) {
      AFrag bf;
      const int bn = tt * 16 + ln;
      bf.h[0] = *(const v8h*)&Bs[cur][bn][hf * 8];
      bf.h[1] = *(const v8h*)&Bs[cur][bn][16 + hf * 8];
      acc[0][tt] = wmma_f16(af[0], bf, acc[0][tt]);
      acc[1][tt] = wmma_f16(af[1], bf, acc[1][tt]);
    }
    __syncthreads();
  }

#pragma unroll
  for (int ms = 0; ms < 2; ++ms)
#pragma unroll
    for (int tt = 0; tt < 4; ++tt) {
      const int c = n0 + tt * 16 + ln;
      const float bcv = bo[c];
#pragma unroll
      for (int v = 0; v < 8; ++v) {
        const int rm = m0 + wave * 32 + ms * 16 + v + 8 * hf;
        out[(size_t)rm * DMOD + c] = acc[ms][tt][v] + bcv;
      }
    }
}

// ---------------------------------------------------------------------------
// Query norms (sum of squares, rank-equivalent to L2 norm)
// ---------------------------------------------------------------------------
__global__ __launch_bounds__(256) void k_norms(const half_t* __restrict__ Qh,
                                               float* __restrict__ nrm)
{
  const int i = blockIdx.x * 256 + threadIdx.x;  // bh*L + l
  const half_t* r = Qh + (size_t)i * HD;
  float s = 0.f;
#pragma unroll
  for (int j = 0; j < HD; j += 8) {
    v8h v = *(const v8h*)(r + j);
#pragma unroll
    for (int e = 0; e < 8; ++e) { float f = (float)v[e]; s += f * f; }
  }
  nrm[i] = s;
}

// ---------------------------------------------------------------------------
// Exact top-u by rank counting, one row per thread (grid 32 x 16).
// Tie-break: lower index first (matches top_k). Ranks unique & dense.
// ---------------------------------------------------------------------------
__global__ __launch_bounds__(256) void k_select(const float* __restrict__ nrm,
                                                int* __restrict__ sel)
{
  __shared__ float ns[LSEQ];
  const int bh = blockIdx.x;
  const float* nb = nrm + (size_t)bh * LSEQ;
  for (int j = threadIdx.x; j < LSEQ; j += 256) ns[j] = nb[j];
  __syncthreads();
  const int l = blockIdx.y * 256 + threadIdx.x;
  const float ni = ns[l];
  int rank = 0;
#pragma unroll 8
  for (int j = 0; j < LSEQ; ++j) {
    const float nj = ns[j];
    rank += ((nj > ni) || (nj == ni && j < l)) ? 1 : 0;
  }
  if (rank < TOPU) sel[bh * SELP + rank] = l;
}

__global__ void k_zero(float4* __restrict__ p) {
  p[blockIdx.x * 256 + threadIdx.x] = make_float4(0.f, 0.f, 0.f, 0.f);
}

// ---------------------------------------------------------------------------
// Sparse attention: one wave per 16-query tile, flash softmax over L,
// 32 keys / step: 4 WMMA (Q@K^T, hd split) + 4 WMMA (P@V).
// ---------------------------------------------------------------------------
__global__ __launch_bounds__(128) void k_attn(
    const half_t* __restrict__ Qh, const half_t* __restrict__ Kh,
    const half_t* __restrict__ Vth, const int* __restrict__ sel,
    half_t* __restrict__ OF)
{
  __shared__ __align__(16) half_t P[4][16][40];   // per-wave P tile, A layout
  const int bh = blockIdx.y;
  const int lane = threadIdx.x & 31, wave = threadIdx.x >> 5;
  const int ln = lane & 15, hf = lane >> 4;
  const int qbase = (blockIdx.x * 4 + wave) * 16;
  const int qi = qbase + ln;
  const int qc = (qi < TOPU) ? qi : (TOPU - 1);
  const int qrow = sel[bh * SELP + qc];

  const half_t* qp = Qh + ((size_t)bh * LSEQ + qrow) * HD;
  AFrag a0, a1;                                   // Q_sel frags, hd 0-31 / 32-63
  a0.h[0] = *(const v8h*)(qp + hf * 8);
  a0.h[1] = *(const v8h*)(qp + 16 + hf * 8);
  a1.h[0] = *(const v8h*)(qp + 32 + hf * 8);
  a1.h[1] = *(const v8h*)(qp + 48 + hf * 8);

  const half_t* Kb = Kh  + (size_t)bh * LSEQ * HD;
  const half_t* Vb = Vth + (size_t)bh * HD * LSEQ;

  float rmax[8], rsum[8];
  v8f acc[4];
#pragma unroll
  for (int v = 0; v < 8; ++v) { rmax[v] = -3.0e38f; rsum[v] = 0.f; }
#pragma unroll
  for (int tt = 0; tt < 4; ++tt) acc[tt] = zero_v8f();

  const float scale = 0.125f;  // 1/sqrt(64)

  for (int l0 = 0; l0 < LSEQ; l0 += 32) {
    v8f s[2];
#pragma unroll
    for (int g = 0; g < 2; ++g) {
      const half_t* kp = Kb + (size_t)(l0 + g * 16 + ln) * HD;
      AFrag b0, b1;
      b0.h[0] = *(const v8h*)(kp + hf * 8);
      b0.h[1] = *(const v8h*)(kp + 16 + hf * 8);
      b1.h[0] = *(const v8h*)(kp + 32 + hf * 8);
      b1.h[1] = *(const v8h*)(kp + 48 + hf * 8);
      s[g] = wmma_f16(a0, b0, zero_v8f());
      s[g] = wmma_f16(a1, b1, s[g]);
    }

    float cf[8];
#pragma unroll
    for (int v = 0; v < 8; ++v) {   // row stats for m = v + 8*hf (per half-wave)
      float t = fmaxf(s[0][v], s[1][v]) * scale;
      t = fmaxf(t, __shfl_xor(t, 1, 32));
      t = fmaxf(t, __shfl_xor(t, 2, 32));
      t = fmaxf(t, __shfl_xor(t, 4, 32));
      t = fmaxf(t, __shfl_xor(t, 8, 32));
      const float nm = fmaxf(rmax[v], t);
      cf[v] = __expf(rmax[v] - nm);
      rmax[v] = nm;
      const float pl = __expf(s[0][v] * scale - nm);
      const float ph = __expf(s[1][v] * scale - nm);
      float rs = pl + ph;
      rs += __shfl_xor(rs, 1, 32);
      rs += __shfl_xor(rs, 2, 32);
      rs += __shfl_xor(rs, 4, 32);
      rs += __shfl_xor(rs, 8, 32);
      rsum[v] = rsum[v] * cf[v] + rs;
      const int q = v + 8 * hf;
      P[wave][q][ln]      = (half_t)pl;           // keys l0+ln
      P[wave][q][ln + 16] = (half_t)ph;           // keys l0+16+ln
    }
#pragma unroll
    for (int tt = 0; tt < 4; ++tt)
#pragma unroll
      for (int v = 0; v < 8; ++v) acc[tt][v] *= cf[v];

    __builtin_amdgcn_wave_barrier();

    AFrag pa;                                     // P as A operand [16q x 32k]
    pa.h[0] = *(const v8h*)&P[wave][ln][hf * 16];
    pa.h[1] = *(const v8h*)&P[wave][ln][hf * 16 + 8];

#pragma unroll
    for (int tt = 0; tt < 4; ++tt) {
      const half_t* vp = Vb + (size_t)(tt * 16 + ln) * LSEQ + l0 + hf * 16;
      AFrag vbf;                                  // B: K(key)=e+16*hf, contig in Vt
      vbf.h[0] = *(const v8h*)vp;
      vbf.h[1] = *(const v8h*)(vp + 8);
      acc[tt] = wmma_f16(pa, vbf, acc[tt]);
    }
    __builtin_prefetch((const void*)(Kb + (size_t)(l0 + 32 + ln) * HD), 0, 0);
    __builtin_prefetch((const void*)(Vb + (size_t)ln * LSEQ + l0 + 32), 0, 0);
  }

  const int bb = bh >> 4, h = bh & 15;
#pragma unroll
  for (int v = 0; v < 8; ++v) {
    const int qg = qbase + v + 8 * hf;
    if (qg < TOPU) {
      const int l = sel[bh * SELP + qg];
      const float inv = 1.0f / rsum[v];
      half_t* dst = OF + ((size_t)bb * LSEQ + l) * DMOD + h * HD + ln;
#pragma unroll
      for (int tt = 0; tt < 4; ++tt) dst[tt * 16] = (half_t)(acc[tt][v] * inv);
    }
  }
}

// ---------------------------------------------------------------------------
extern "C" void kernel_launch(void* const* d_in, const int* in_sizes, int n_in,
                              void* d_out, int out_size, void* d_ws, size_t ws_size,
                              hipStream_t stream) {
  (void)in_sizes; (void)n_in; (void)out_size; (void)ws_size;
  const float* x  = (const float*)d_in[0];
  const float* Wq = (const float*)d_in[1];
  const float* bq = (const float*)d_in[2];
  const float* Wk = (const float*)d_in[3];
  const float* bk = (const float*)d_in[4];
  const float* Wv = (const float*)d_in[5];
  const float* bv = (const float*)d_in[6];
  const float* Wo = (const float*)d_in[7];
  const float* bo = (const float*)d_in[8];
  float* out = (float*)d_out;

  char* ws = (char*)d_ws;
  const size_t qkv_bytes = (size_t)BHT * LSEQ * HD * sizeof(half_t);   // 16 MB
  const size_t w_bytes   = (size_t)DMOD * DMOD * sizeof(half_t);       //  2 MB
  half_t* Qh  = (half_t*)ws;  ws += qkv_bytes;
  half_t* Kh  = (half_t*)ws;  ws += qkv_bytes;
  half_t* Vth = (half_t*)ws;  ws += qkv_bytes;
  half_t* OF  = (half_t*)ws;  ws += (size_t)BDIM * LSEQ * DMOD * sizeof(half_t);
  half_t* xh  = (half_t*)ws;  ws += (size_t)BDIM * LSEQ * DMOD * sizeof(half_t);
  half_t* Whq = (half_t*)ws;  ws += w_bytes;
  half_t* Whk = (half_t*)ws;  ws += w_bytes;
  half_t* Whv = (half_t*)ws;  ws += w_bytes;
  half_t* Who = (half_t*)ws;  ws += w_bytes;
  float*  nrm = (float*)ws;   ws += (size_t)BHT * LSEQ * sizeof(float);
  int*    sel = (int*)ws;     ws += (size_t)BHT * SELP * sizeof(int);

  k_cvt<<<(BDIM * LSEQ * DMOD) / 2048, 256, 0, stream>>>(x, xh);
  k_cvt<<<(DMOD * DMOD) / 2048, 256, 0, stream>>>(Wq, Whq);
  k_cvt<<<(DMOD * DMOD) / 2048, 256, 0, stream>>>(Wk, Whk);
  k_cvt<<<(DMOD * DMOD) / 2048, 256, 0, stream>>>(Wv, Whv);
  k_cvt<<<(DMOD * DMOD) / 2048, 256, 0, stream>>>(Wo, Who);

  k_proj_qkv<<<dim3(DMOD / 64, (BDIM * LSEQ) / 128, 3), 128, 0, stream>>>(
      xh, Whq, Whk, Whv, bq, bk, bv, Qh, Kh, Vth);
  k_norms<<<(BHT * LSEQ) / 256, 256, 0, stream>>>(Qh, nrm);
  k_select<<<dim3(BHT, LSEQ / 256), 256, 0, stream>>>(nrm, sel);
  k_zero<<<4096, 256, 0, stream>>>((float4*)OF);   // 16 MB out_full -> 0
  k_attn<<<dim3(13, BHT), 128, 0, stream>>>(Qh, Kh, Vth, sel, OF);
  k_out_proj<<<dim3(DMOD / 64, (BDIM * LSEQ) / 128), 128, 0, stream>>>(
      OF, Who, bo, out);
}
